// MultiHeadAttention_31327491457439
// MI455X (gfx1250) — compile-verified, matrix operand order checked by reading, not run on hardware
//
#include <hip/hip_runtime.h>

// ---------------------------------------------------------------------------
// MHA forward for MI455X (gfx1250, wave32, WMMA, async-LDS + TDM staging).
// ---------------------------------------------------------------------------

namespace {
constexpr int kB   = 2;
constexpr int kS   = 2048;
constexpr int kHID = 2048;
constexpr int kH   = 16;
constexpr int kD   = 128;              // head dim
constexpr int kM   = kB * kS;          // 4096 total rows
constexpr float kScale = 11.313708498984761f;  // sqrt(128); reference MULTIPLIES q
}

typedef __attribute__((ext_vector_type(16))) __bf16 v16bf;
typedef __attribute__((ext_vector_type(8)))  __bf16 v8bf;
typedef __attribute__((ext_vector_type(8)))  float  v8f;
typedef __attribute__((ext_vector_type(4)))  unsigned int v4u;
typedef __attribute__((ext_vector_type(8)))  int v8i;
typedef __attribute__((ext_vector_type(4)))  int v4i;

__device__ __forceinline__ v8f vzero8() {
  v8f a;
#pragma unroll
  for (int i = 0; i < 8; ++i) a[i] = 0.0f;
  return a;
}

// One WMMA: D = A(16x32 bf16) * B(32x16 bf16) + C(16x16 f32)
__device__ __forceinline__ v8f wmma_bf16(v16bf a, v16bf b, v8f c) {
  return __builtin_amdgcn_wmma_f32_16x16x32_bf16(
      /*neg_a=*/false, a, /*neg_b=*/false, b,
      /*c_mod=*/(short)0, c, /*reuse_a=*/false, /*reuse_b=*/false);
}

// Load a 16x32 fragment from a row-major bf16 matrix [rows x ld].
// Per-lane ISA layout: row = r0 + lane%16; elems 0..7 at k = k0 + 8*(lane/16),
// elems 8..15 at k+16.  Two 16-byte loads per lane (global or LDS).
__device__ __forceinline__ v16bf load_frag(const __bf16* base, int ld,
                                           int r0, int k0, int lane) {
  const int r  = r0 + (lane & 15);
  const int kb = k0 + ((lane >> 4) << 3);
  const __bf16* p = base + (size_t)r * ld + kb;
  v8bf lo = *(const v8bf*)(p);
  v8bf hi = *(const v8bf*)(p + 16);
  v16bf f;
#pragma unroll
  for (int i = 0; i < 8; ++i) { f[i] = lo[i]; f[i + 8] = hi[i]; }
  return f;
}

// ---- CDNA5 async global->LDS copy (ASYNCcnt-tracked) ----------------------
__device__ __forceinline__ void async_ld_b128(unsigned lds_byte_off, const void* g) {
  asm volatile("global_load_async_to_lds_b128 %0, %1, off"
               :: "v"(lds_byte_off), "v"(g) : "memory");
}
__device__ __forceinline__ void wait_async0() {
  asm volatile("s_wait_asynccnt 0x0" ::: "memory");
}

// ---- CDNA5 Tensor Data Mover descriptors (ISA ch.8 D# layout) -------------
// Group 0: [1:0]=count=1, [63:32]=lds_addr, [120:64]=global_addr, [127:126]=2.
__device__ __forceinline__ v4u tdm_group0(unsigned lds_addr, const void* gaddr) {
  unsigned long long ga = (unsigned long long)gaddr;
  v4u g;
  g[0] = 1u;
  g[1] = lds_addr;
  g[2] = (unsigned)(ga & 0xffffffffu);
  g[3] = (unsigned)((ga >> 32) & 0x01ffffffu) | (2u << 30);
  return g;
}
// Group 1 for a 2D tile of 8-byte elements: tile_d0 x tile_d1 (tile_d1=0 -> 1D),
// row stride = stride0 elems.  data_size=3 (8B); pad/iterate/multicast off.
__device__ __forceinline__ v8i tdm_group1(unsigned tile_d0, unsigned tile_d1,
                                          unsigned long long stride0,
                                          unsigned tensor_d0, unsigned tensor_d1) {
  v8i g;
  g[0] = (int)(3u << 16);                                        // data_size=8B
  g[1] = (int)((tensor_d0 & 0xffffu) << 16);                     // tensor_dim0 lo
  g[2] = (int)((tensor_d0 >> 16) | ((tensor_d1 & 0xffffu) << 16));
  g[3] = (int)((tensor_d1 >> 16) | (tile_d0 << 16));             // tile_dim0
  g[4] = (int)(tile_d1 & 0xffffu);                               // tile_dim1
  g[5] = (int)(stride0 & 0xffffffffu);                           // dim0_stride lo
  g[6] = (int)((stride0 >> 32) & 0xffffu);                       // dim0_stride hi
  g[7] = 0;
  return g;
}

// ---------------------------------------------------------------------------
// Kernel 0: fp32 -> bf16 elementwise
// ---------------------------------------------------------------------------
__global__ void cvt_bf16_kernel(const float* __restrict__ in,
                                __bf16* __restrict__ out, int n) {
  int i = blockIdx.x * blockDim.x + threadIdx.x;
  if (i < n) out[i] = (__bf16)in[i];
}

// Kernel 1: fp32 [rows x cols] -> bf16 transposed [cols x rows]
__global__ void transpose_cvt_kernel(const float* __restrict__ in,
                                     __bf16* __restrict__ out,
                                     int rows, int cols) {
  int i = blockIdx.x * blockDim.x + threadIdx.x;
  if (i < rows * cols) {
    int r = i / cols, c = i % cols;
    out[(size_t)c * rows + r] = (__bf16)in[i];
  }
}

// ---------------------------------------------------------------------------
// Kernel 2: GEMM  C = A(bf16, M x 2048) @ Bt^T + bias, various epilogues.
//   Bt[n][k] = W[k][n]  (bf16, N x K) so B-fragments load contiguously.
//   MODE 0: Q -> bf16 [B,H,S,D] * sqrt(D);  MODE 1: K -> bf16 [B,H,S,D]
//   MODE 2: V -> bf16 [B,H,D,S];            MODE 3: fp32 [M, 2048]
// ---------------------------------------------------------------------------
template <int MODE>
__global__ __launch_bounds__(128, 2)
void gemm_proj_kernel(const __bf16* __restrict__ A,
                      const __bf16* __restrict__ Bt,
                      const float* __restrict__ bias,
                      void* __restrict__ out, float scale) {
  const int lane = threadIdx.x & 31;
  const int wave = threadIdx.x >> 5;
  const int m0 = blockIdx.x * 32;
  const int n0 = blockIdx.y * 256 + wave * 64;

  v8f acc[2][4];
#pragma unroll
  for (int r = 0; r < 2; ++r)
#pragma unroll
    for (int j = 0; j < 4; ++j) acc[r][j] = vzero8();

  for (int k0 = 0; k0 < kHID; k0 += 32) {
    v16bf a0 = load_frag(A, kHID, m0,      k0, lane);
    v16bf a1 = load_frag(A, kHID, m0 + 16, k0, lane);
#pragma unroll
    for (int j = 0; j < 4; ++j) {
      v16bf b = load_frag(Bt, kHID, n0 + j * 16, k0, lane);
      acc[0][j] = wmma_bf16(a0, b, acc[0][j]);
      acc[1][j] = wmma_bf16(a1, b, acc[1][j]);
    }
  }

  const int grp = lane >> 4;
  const int nl  = lane & 15;
#pragma unroll
  for (int r = 0; r < 2; ++r) {
#pragma unroll
    for (int j = 0; j < 4; ++j) {
#pragma unroll
      for (int i = 0; i < 8; ++i) {
        const int m = m0 + r * 16 + i + 8 * grp;
        const int n = n0 + j * 16 + nl;
        const float v = (acc[r][j][i] + bias[n]) * scale;
        if (MODE == 0 || MODE == 1) {
          const int bb = m / kS, ss = m % kS, hh = n / kD, dd = n % kD;
          ((__bf16*)out)[(((size_t)bb * kH + hh) * kS + ss) * kD + dd] = (__bf16)v;
        } else if (MODE == 2) {
          const int bb = m / kS, ss = m % kS, hh = n / kD, dd = n % kD;
          ((__bf16*)out)[(((size_t)bb * kH + hh) * kD + dd) * kS + ss] = (__bf16)v;
        } else {
          ((float*)out)[(size_t)m * kHID + n] = v;
        }
      }
    }
  }
}

// ---------------------------------------------------------------------------
// Kernel 3: flash attention, double-buffered K (async-LDS) + V (TDM) staging.
//   Q,K: bf16 [B,H,S,D]; Vt: bf16 [B,H,D,S]; bias: fp32 [B,H,S,S];
//   ctx: bf16 [B,S,H*D].
// Block = 128 threads (4 waves) = 64 q-rows of one (b,h); keys in steps of 32.
// LDS map (single static shared block => LDS offset 0):
//   [0,8K)    K tile buf0 (32 keys x 128 d, ld=128)   [8K,16K)  K tile buf1
//   [16K,24K) V tile buf0 (128 d x 32 keys, ld=32)    [24K,32K) V tile buf1
//   [32K,36K) per-wave P tiles (16x32)
// ---------------------------------------------------------------------------
#define SMEM_BYTES 36864
#define KOFF0 0u
#define KOFF1 8192u
#define VOFF0 16384u
#define VOFF1 24576u
#define POFF  32768u

__global__ __launch_bounds__(128, 1)
void attn_kernel(const __bf16* __restrict__ Q,
                 const __bf16* __restrict__ K,
                 const __bf16* __restrict__ Vt,
                 const float* __restrict__ bias,
                 __bf16* __restrict__ ctx) {
  __shared__ __align__(16) char smem_raw[SMEM_BYTES];
  __bf16* sbase = (__bf16*)smem_raw;

  const int tid  = threadIdx.x;
  const int lane = tid & 31;
  const int wave = tid >> 5;
  const int bh   = blockIdx.y;
  const int b    = bh / kH;
  const int h    = bh % kH;
  const int q0   = blockIdx.x * 64 + wave * 16;

  const __bf16* Qb = Q  + (size_t)bh * kS * kD;
  const __bf16* Kb = K  + (size_t)bh * kS * kD;
  const __bf16* Vb = Vt + (size_t)bh * kD * kS;
  const float*  Bb = bias + (size_t)bh * kS * kS;

  // K tile: contiguous 8KB, per-thread async b128 x4.
  auto stageK = [&](int buf, int kt) {
    const unsigned ko = (buf ? KOFF1 : KOFF0) + (unsigned)tid * 64u;
    const char* kg = (const char*)(Kb + (size_t)kt * kD) + tid * 64;
#pragma unroll
    for (int j = 0; j < 4; ++j) async_ld_b128(ko + j * 16u, kg + j * 16);
  };
  // V tile: 2D strided (128 rows x 64B, row stride 2*kS bytes) via TDM, wave 0.
  auto stageV = [&](int buf, int kt) {
    if (wave == 0) {
      const v4i z4 = {0, 0, 0, 0};
      const v8i z8 = {0, 0, 0, 0, 0, 0, 0, 0};
      __builtin_amdgcn_tensor_load_to_lds(
          tdm_group0(buf ? VOFF1 : VOFF0, (const void*)(Vb + kt)),
          tdm_group1(/*tile_d0=*/8, /*tile_d1=*/128, /*stride0=*/kS * 2 / 8,
                     /*tensor_d0=*/1u << 20, /*tensor_d1=*/1u << 20),
          z4, z4, z8, 0);
    }
  };

  // Resident Q fragments: 16 x 128 = 4 K-chunks of 32.
  v16bf qf[4];
#pragma unroll
  for (int c = 0; c < 4; ++c) qf[c] = load_frag(Qb, kD, q0, c * 32, lane);

  v8f o[8];
#pragma unroll
  for (int nn = 0; nn < 8; ++nn) o[nn] = vzero8();
  float mx[8], ls[8];
#pragma unroll
  for (int i = 0; i < 8; ++i) { mx[i] = -3.0e38f; ls[i] = 0.0f; }

  const int grp = lane >> 4;
  const int nl  = lane & 15;
  __bf16* pt = sbase + (POFF / 2) + wave * 512;   // 16x32 bf16 per wave

  // Preload first tile.
  stageK(0, 0);
  stageV(0, 0);
  wait_async0();
  if (wave == 0) __builtin_amdgcn_s_wait_tensorcnt(0);
  __syncthreads();

  for (int kt = 0; kt < kS; kt += 32) {
    const int buf = (kt >> 5) & 1;
    const __bf16* sK = sbase + (buf ? KOFF1 : KOFF0) / 2;
    const __bf16* sV = sbase + (buf ? VOFF1 : VOFF0) / 2;

    if (kt + 32 < kS) {
      stageK(buf ^ 1, kt + 32);                     // async DMA next K tile
      stageV(buf ^ 1, kt + 32);                     // TDM next V tile
      __builtin_prefetch(Bb + (size_t)q0 * kS + kt + 32, 0, 1);
    }

    // ---- S = Q K^T: preload all 8 B-fragments, then 8 WMMAs ----
    v16bf kf[8];
#pragma unroll
    for (int c = 0; c < 4; ++c) {
      kf[c]     = load_frag(sK, kD, 0,  c * 32, lane);  // keys kt..kt+15
      kf[4 + c] = load_frag(sK, kD, 16, c * 32, lane);  // keys kt+16..kt+31
    }
    v8f s0 = vzero8(), s1 = vzero8();
#pragma unroll
    for (int c = 0; c < 4; ++c) {
      s0 = wmma_bf16(qf[c], kf[c],     s0);
      s1 = wmma_bf16(qf[c], kf[4 + c], s1);
    }

    // ---- bias add + online softmax (rows live in VGPR index i) ----
#pragma unroll
    for (int i = 0; i < 8; ++i) {
      const int m = q0 + i + 8 * grp;
      s0[i] += Bb[(size_t)m * kS + kt + nl];
      s1[i] += Bb[(size_t)m * kS + kt + 16 + nl];

      float t = fmaxf(s0[i], s1[i]);
      t = fmaxf(t, __shfl_xor(t, 1));
      t = fmaxf(t, __shfl_xor(t, 2));
      t = fmaxf(t, __shfl_xor(t, 4));
      t = fmaxf(t, __shfl_xor(t, 8));
      const float mnew = fmaxf(mx[i], t);
      const float corr = __expf(mx[i] - mnew);
      mx[i] = mnew;

      const float p0 = __expf(s0[i] - mnew);
      const float p1 = __expf(s1[i] - mnew);
      s0[i] = p0; s1[i] = p1;
      float rs = p0 + p1;
      rs += __shfl_xor(rs, 1);
      rs += __shfl_xor(rs, 2);
      rs += __shfl_xor(rs, 4);
      rs += __shfl_xor(rs, 8);
      ls[i] = ls[i] * corr + rs;

#pragma unroll
      for (int nn = 0; nn < 8; ++nn) o[nn][i] *= corr;
    }

    // ---- P (16x32) -> per-wave LDS tile -> A-fragment ----
#pragma unroll
    for (int i = 0; i < 8; ++i) {
      pt[(i + 8 * grp) * 32 + nl]      = (__bf16)s0[i];
      pt[(i + 8 * grp) * 32 + 16 + nl] = (__bf16)s1[i];
    }
    __syncthreads();                                   // barrier #1
    v16bf pa = load_frag(pt, 32, 0, 0, lane);

    // ---- O += P V : 8 output chunks, V-fragments preloaded in halves ----
#pragma unroll
    for (int half = 0; half < 2; ++half) {
      v16bf vf[4];
#pragma unroll
      for (int j = 0; j < 4; ++j)
        vf[j] = load_frag(sV, 32, (half * 4 + j) * 16, 0, lane);
#pragma unroll
      for (int j = 0; j < 4; ++j)
        o[half * 4 + j] = wmma_bf16(pa, vf[j], o[half * 4 + j]);
    }

    wait_async0();                                     // my next K DMAs done
    if (wave == 0) __builtin_amdgcn_s_wait_tensorcnt(0);  // V TDM done
    __syncthreads();                                   // barrier #2
  }

  // ---- normalize and write ctx [B,S,HID] ----
  float inv[8];
#pragma unroll
  for (int i = 0; i < 8; ++i) inv[i] = 1.0f / ls[i];
#pragma unroll
  for (int nn = 0; nn < 8; ++nn) {
#pragma unroll
    for (int i = 0; i < 8; ++i) {
      const int ss = q0 + i + 8 * grp;
      const int dd = nn * 16 + nl;
      ctx[((size_t)b * kS + ss) * kHID + h * kD + dd] = (__bf16)(o[nn][i] * inv[i]);
    }
  }
}

// ---------------------------------------------------------------------------
// Host launcher
// ---------------------------------------------------------------------------
extern "C" void kernel_launch(void* const* d_in, const int* in_sizes, int n_in,
                              void* d_out, int out_size, void* d_ws, size_t ws_size,
                              hipStream_t stream) {
  const float* x    = (const float*)d_in[0];
  const float* bias = (const float*)d_in[1];
  const float* Wq   = (const float*)d_in[2];
  const float* bq   = (const float*)d_in[3];
  const float* Wk   = (const float*)d_in[4];
  const float* bk   = (const float*)d_in[5];
  const float* Wv   = (const float*)d_in[6];
  const float* bv   = (const float*)d_in[7];
  const float* Wo   = (const float*)d_in[8];
  const float* bo   = (const float*)d_in[9];
  float* out = (float*)d_out;

  // Workspace layout (bf16 scratch), ~117 MB total.
  char* ws = (char*)d_ws;
  const size_t actBytes = (size_t)kM * kHID * sizeof(__bf16);     // 16 MB
  const size_t wBytes   = (size_t)kHID * kHID * sizeof(__bf16);   // 8 MB
  __bf16* xbf  = (__bf16*)ws;             ws += actBytes;
  __bf16* wqT  = (__bf16*)ws;             ws += wBytes;
  __bf16* wkT  = (__bf16*)ws;             ws += wBytes;
  __bf16* wvT  = (__bf16*)ws;             ws += wBytes;
  __bf16* woT  = (__bf16*)ws;             ws += wBytes;
  __bf16* qh   = (__bf16*)ws;             ws += actBytes;   // [B,H,S,D], pre-scaled
  __bf16* kh   = (__bf16*)ws;             ws += actBytes;   // [B,H,S,D]
  __bf16* vth  = (__bf16*)ws;             ws += actBytes;   // [B,H,D,S]
  __bf16* ctxb = (__bf16*)ws;             ws += actBytes;   // [B,S,HID]

  // 0) conversions
  {
    const int n = kM * kHID;
    cvt_bf16_kernel<<<(n + 255) / 256, 256, 0, stream>>>(x, xbf, n);
    const int nw = kHID * kHID;
    dim3 g((nw + 255) / 256);
    transpose_cvt_kernel<<<g, 256, 0, stream>>>(Wq, wqT, kHID, kHID);
    transpose_cvt_kernel<<<g, 256, 0, stream>>>(Wk, wkT, kHID, kHID);
    transpose_cvt_kernel<<<g, 256, 0, stream>>>(Wv, wvT, kHID, kHID);
    transpose_cvt_kernel<<<g, 256, 0, stream>>>(Wo, woT, kHID, kHID);
  }

  // 1) QKV projections (WMMA)
  dim3 gGemm(kM / 32, kHID / 256);
  gemm_proj_kernel<0><<<gGemm, 128, 0, stream>>>(xbf, wqT, bq, qh,  kScale);
  gemm_proj_kernel<1><<<gGemm, 128, 0, stream>>>(xbf, wkT, bk, kh,  1.0f);
  gemm_proj_kernel<2><<<gGemm, 128, 0, stream>>>(xbf, wvT, bv, vth, 1.0f);

  // 2) attention (WMMA flash, async + TDM staging)
  dim3 gAttn(kS / 64, kB * kH);
  attn_kernel<<<gAttn, 128, 0, stream>>>(qh, kh, vth, bias, ctxb);

  // 3) output projection (WMMA, fp32 out)
  gemm_proj_kernel<3><<<gGemm, 128, 0, stream>>>(ctxb, woT, bo, out, 1.0f);
}
